// StandardAttention_32744830665501
// MI455X (gfx1250) — matrix operand order repaired and checked by reading, not compiled, not run on hardware
//
#include <hip/hip_runtime.h>
#include <hip/hip_bf16.h>
#include <math.h>
#include <stdint.h>

// ---------------------------------------------------------------------------
// Problem constants (reference: B=2, S=2048, DIM=1024, H=16, Dh=64)
// ---------------------------------------------------------------------------
#define BB      2
#define SS      2048
#define DIMM    1024
#define HH      16
#define DHH     64
#define MTOT    (BB * SS)          // 4096 rows of x
#define NQKV    (3 * DIMM)         // 3072 fused qkv output features

typedef __attribute__((ext_vector_type(16))) _Float16 v16h;
typedef __attribute__((ext_vector_type(8)))  _Float16 v8h;
typedef __attribute__((ext_vector_type(8)))  float    v8f;
typedef __attribute__((ext_vector_type(4)))  unsigned int v4u;
typedef __attribute__((ext_vector_type(8)))  int      v8i;
typedef __attribute__((ext_vector_type(4)))  int      v4i;

// Build a 16x32 (f16) WMMA A/B fragment from two 16-byte chunks.
static __device__ inline v16h make_frag(const _Float16* lo_p, const _Float16* hi_p) {
  v8h lo = *(const v8h*)lo_p;
  v8h hi = *(const v8h*)hi_p;
  v16h r;
#pragma unroll
  for (int i = 0; i < 8; ++i) { r[i] = lo[i]; r[i + 8] = hi[i]; }
  return r;
}

// ---------------------------------------------------------------------------
// TDM: 2D tile load Global -> LDS.  D# built per CDNA5 ISA:
//   group0: [1:0]=count=1, [63:32]=lds byte addr, [120:64]=global addr,
//           [127:126]=type=2
//   group1: [17:16]=data_size(1 => 2B), [79:48]=tensor_dim0, [111:80]=tensor_dim1,
//           [127:112]=tile_dim0, [143:128]=tile_dim1, [207:160]=dim0_stride
// Tile rows are written back-to-back into LDS => row-major [t1][t0] f16 tile.
// ---------------------------------------------------------------------------
static __device__ inline void tdm_load_2d_f16(const _Float16* gtile, uint32_t lds_byte,
                                              uint32_t dim0, uint32_t dim1,
                                              uint32_t t0,  uint32_t t1,
                                              uint32_t stride0) {
  uint64_t ga = (uint64_t)(uintptr_t)gtile;
  v4u g0;
  g0[0] = 1u;                                             // count=1, load, user
  g0[1] = lds_byte;                                       // LDS byte address
  g0[2] = (uint32_t)ga;                                   // global_addr[31:0]
  g0[3] = (uint32_t)((ga >> 32) & 0x01FFFFFFu) | (2u << 30);  // addr[56:32] | type=2
  v8i g1;
  g1[0] = (int)(1u << 16);                                // data_size = 2 bytes
  g1[1] = (int)((dim0 & 0xFFFFu) << 16);                  // tensor_dim0[15:0]
  g1[2] = (int)((dim0 >> 16) | ((dim1 & 0xFFFFu) << 16)); // dim0[31:16] | dim1[15:0]
  g1[3] = (int)((dim1 >> 16) | (t0 << 16));               // dim1[31:16] | tile_dim0
  g1[4] = (int)(t1 & 0xFFFFu);                            // tile_dim1 (tile_dim2=0)
  g1[5] = (int)stride0;                                   // dim0_stride[31:0]
  g1[6] = 0;                                              // stride hi | dim1_stride lo
  g1[7] = 0;
  v4i z4 = {0, 0, 0, 0};
#if __clang_major__ >= 23
  v8i z8 = {0, 0, 0, 0, 0, 0, 0, 0};
  __builtin_amdgcn_tensor_load_to_lds(g0, g1, z4, z4, z8, 0);
#else
  __builtin_amdgcn_tensor_load_to_lds(g0, g1, z4, z4, 0);
#endif
}

// ---------------------------------------------------------------------------
// K0: f32 -> f16 convert
// ---------------------------------------------------------------------------
__global__ __launch_bounds__(256) void cvt_f32_to_f16(const float* __restrict__ src,
                                                      _Float16* __restrict__ dst, int n) {
  int i = blockIdx.x * 256 + threadIdx.x;
  if (i < n) dst[i] = (_Float16)src[i];
}

// ---------------------------------------------------------------------------
// K1/K4: C[M,N] = A[M,K] @ W[N,K]^T + bias[N]   (A,W f16; C f32)
// Block: 256 thr = 8 waves (2 x 4).  Wave tile: 32(M) x 64(N) = 2x4 WMMA tiles.
// Block tile: 64(M) x 256(N).  K-step 32, TDM double-buffered through LDS.
// Grid: (N/256, M/64).
// ---------------------------------------------------------------------------
__global__ __launch_bounds__(256) void gemm_f16_wmma(const _Float16* __restrict__ A,
                                                     const _Float16* __restrict__ W,
                                                     const float* __restrict__ bias,
                                                     float* __restrict__ C,
                                                     int M, int N, int K) {
  __shared__ _Float16 ldsA[2][64 * 32];    //  8 KB: A tile, row-major [64][32]
  __shared__ _Float16 ldsW[2][256 * 32];   // 32 KB: W tile, row-major [256][32]

  const int lane = threadIdx.x & 31;
  const int wave = threadIdx.x >> 5;
  const int wm = wave >> 2;                  // 0..1
  const int wn = wave & 3;                   // 0..3
  const int mblk = blockIdx.y * 64;
  const int nblk = blockIdx.x * 256;
  const int lm = lane & 15;
  const int g  = lane >> 4;

  const uint32_t ldsA_b[2] = { (uint32_t)(uintptr_t)(void*)&ldsA[0][0],
                               (uint32_t)(uintptr_t)(void*)&ldsA[1][0] };
  const uint32_t ldsW_b[2] = { (uint32_t)(uintptr_t)(void*)&ldsW[0][0],
                               (uint32_t)(uintptr_t)(void*)&ldsW[1][0] };

  v8f acc[2][4];
#pragma unroll
  for (int mt = 0; mt < 2; ++mt)
#pragma unroll
    for (int nt = 0; nt < 4; ++nt)
#pragma unroll
      for (int r = 0; r < 8; ++r) acc[mt][nt][r] = 0.0f;

  const int nsteps = K >> 5;                 // K/32
  if (wave == 0) {                           // prologue: fill buffer 0
    tdm_load_2d_f16(A + (size_t)mblk * K, ldsA_b[0], (uint32_t)K, (uint32_t)(M - mblk),
                    32u, 64u, (uint32_t)K);
    tdm_load_2d_f16(W + (size_t)nblk * K, ldsW_b[0], (uint32_t)K, (uint32_t)(N - nblk),
                    32u, 256u, (uint32_t)K);
  }

  for (int i = 0; i < nsteps; ++i) {
    const int cur = i & 1;
    if (wave == 0) {
      if (i + 1 < nsteps) {                  // kick off next tile, then wait for cur
        const int k = (i + 1) << 5;
        tdm_load_2d_f16(A + (size_t)mblk * K + k, ldsA_b[cur ^ 1], (uint32_t)K,
                        (uint32_t)(M - mblk), 32u, 64u, (uint32_t)K);
        tdm_load_2d_f16(W + (size_t)nblk * K + k, ldsW_b[cur ^ 1], (uint32_t)K,
                        (uint32_t)(N - nblk), 32u, 256u, (uint32_t)K);
        __builtin_amdgcn_s_wait_tensorcnt(2);   // in-order TDM: cur's 2 loads done
      } else {
        __builtin_amdgcn_s_wait_tensorcnt(0);
      }
    }
    __syncthreads();                         // publish cur buffer to all 8 waves

    // A fragments: row (wm*32 + mt*16 + lm), chunks at k-local 8g / 16+8g
    v16h af[2];
#pragma unroll
    for (int mt = 0; mt < 2; ++mt) {
      const _Float16* ap = &ldsA[cur][(wm * 32 + mt * 16 + lm) * 32];
      af[mt] = make_frag(ap + 8 * g, ap + 16 + 8 * g);
    }
    // B fragments: W row (wn*64 + nt*16 + lm), chunks at k-local 8g / 16+8g
    v16h bf[4];
#pragma unroll
    for (int nt = 0; nt < 4; ++nt) {
      const _Float16* wp = &ldsW[cur][(wn * 64 + nt * 16 + lm) * 32];
      bf[nt] = make_frag(wp + 8 * g, wp + 16 + 8 * g);
    }
#pragma unroll
    for (int mt = 0; mt < 2; ++mt)
#pragma unroll
      for (int nt = 0; nt < 4; ++nt)
        acc[mt][nt] = __builtin_amdgcn_wmma_f32_16x16x32_f16(
            false, af[mt], false, bf[nt], (short)0, acc[mt][nt], false, false);

    __syncthreads();                         // all reads done before TDM overwrites
  }

  // D layout: element r -> row m = r + 8g, col n = lm
#pragma unroll
  for (int mt = 0; mt < 2; ++mt)
#pragma unroll
    for (int nt = 0; nt < 4; ++nt)
#pragma unroll
      for (int r = 0; r < 8; ++r) {
        int row = mblk + wm * 32 + mt * 16 + r + 8 * g;
        int col = nblk + wn * 64 + nt * 16 + lm;
        C[(size_t)row * N + col] = acc[mt][nt][r] + bias[col];
      }
}

// ---------------------------------------------------------------------------
// K2: RoPE + L2-norm on q,k; transpose v.  One wave per (b,s,h); lane owns
// the interleaved pair d = 2*lane, 2*lane+1.
//   qkv : f32 [B*S, 3072]  ->  qh,kh : f16 [B,H,S,64] ; vt : f16 [B,H,64,S]
// ---------------------------------------------------------------------------
__global__ __launch_bounds__(256) void rope_norm_kernel(const float* __restrict__ qkv,
                                                        _Float16* __restrict__ qh,
                                                        _Float16* __restrict__ kh,
                                                        _Float16* __restrict__ vt) {
  const int lane = threadIdx.x & 31;
  const int wave = threadIdx.x >> 5;
  const int idx = blockIdx.x * 8 + wave;     // (b, s, h) flat, B*S*H = 65536
  const int b   = idx >> 15;                 // / (S*H)
  const int rem = idx & 32767;
  const int s   = rem >> 4;                  // / H
  const int h   = rem & 15;
  const int d0  = lane * 2;

  const float* base = qkv + (size_t)(b * SS + s) * NQKV + h * DHH + d0;
  float q0 = base[0],            q1 = base[1];
  float k0 = base[DIMM],         k1 = base[DIMM + 1];
  float v0 = base[2 * DIMM],     v1 = base[2 * DIMM + 1];

  // freq_i = 10000^(-(2i)/64), i = d0/2 = lane  ->  exponent -(d0)/64
  float f   = __powf(10000.0f, -(float)d0 / (float)DHH);
  float ang = (float)s * f;
  float c = __cosf(ang), sn = __sinf(ang);
  // interleaved rotate_half: out[2i]=-x[2i+1], out[2i+1]=x[2i]
  float qr0 = q0 * c - q1 * sn, qr1 = q1 * c + q0 * sn;
  float kr0 = k0 * c - k1 * sn, kr1 = k1 * c + k0 * sn;

  float qs = qr0 * qr0 + qr1 * qr1;
  float ks = kr0 * kr0 + kr1 * kr1;
#pragma unroll
  for (int m = 16; m >= 1; m >>= 1) {
    qs += __shfl_xor(qs, m, 32);
    ks += __shfl_xor(ks, m, 32);
  }
  float qi = 1.0f / sqrtf(qs + 1e-6f);
  float ki = 1.0f / sqrtf(ks + 1e-6f);

  size_t qo = ((size_t)(b * HH + h) * SS + s) * DHH + d0;
  qh[qo]     = (_Float16)(qr0 * qi);
  qh[qo + 1] = (_Float16)(qr1 * qi);
  kh[qo]     = (_Float16)(kr0 * ki);
  kh[qo + 1] = (_Float16)(kr1 * ki);

  size_t vo = ((size_t)(b * HH + h) * DHH + d0) * SS + s;
  vt[vo]      = (_Float16)v0;
  vt[vo + SS] = (_Float16)v1;
}

// ---------------------------------------------------------------------------
// K3: causal flash attention with WMMA.  Block = 128 thr = 4 waves; each wave
// owns 16 query rows of one (b,h); streams keys 32 at a time with online
// softmax.  Output: ctx f16 [B*S, DIM] ready for the Wo GEMM.
// (No block barriers: waves have different trip counts; LDS use is wave-local.)
// ---------------------------------------------------------------------------
__global__ __launch_bounds__(128) void flash_attn_wmma(const _Float16* __restrict__ qh,
                                                       const _Float16* __restrict__ kh,
                                                       const _Float16* __restrict__ vt,
                                                       const float* __restrict__ logit_scale,
                                                       _Float16* __restrict__ ctxh) {
  __shared__ _Float16 pbuf[4][16][40];       // per-wave 16x32 P tile, pitch 80B (16B aligned)
  const int lane = threadIdx.x & 31;
  const int wave = threadIdx.x >> 5;
  const int bh   = blockIdx.x;               // 0..31
  const int b    = bh >> 4, h = bh & 15;
  const int q0   = blockIdx.y * 64 + wave * 16;
  const int lm   = lane & 15;
  const int g    = lane >> 4;
  const float sc = 0.125f * __expf(logit_scale[0]);   // Dh^-0.5 * exp(logit_scale)

  const _Float16* Q  = qh + (size_t)bh * SS * DHH;
  const _Float16* Km = kh + (size_t)bh * SS * DHH;
  const _Float16* Vt = vt + (size_t)bh * DHH * SS;

  // Q A-fragments for d-chunks [0,32) and [32,64)
  v16h aq[2];
#pragma unroll
  for (int c = 0; c < 2; ++c) {
    const _Float16* qp = Q + (size_t)(q0 + lm) * DHH + 32 * c;
    aq[c] = make_frag(qp + 8 * g, qp + 16 + 8 * g);
  }

  v8f out[4];
  float mrow[8], lrow[8];
#pragma unroll
  for (int dt = 0; dt < 4; ++dt)
#pragma unroll
    for (int r = 0; r < 8; ++r) out[dt][r] = 0.0f;
#pragma unroll
  for (int r = 0; r < 8; ++r) { mrow[r] = -INFINITY; lrow[r] = 0.0f; }

  const int qlast = q0 + 15;
  for (int kb = 0; kb <= qlast; kb += 32) {
    // ---- scores: S0 = Q @ K[kb..kb+15]^T, S1 = Q @ K[kb+16..kb+31]^T ----
    v8f s0, s1;
#pragma unroll
    for (int r = 0; r < 8; ++r) { s0[r] = 0.0f; s1[r] = 0.0f; }
#pragma unroll
    for (int c = 0; c < 2; ++c) {
      const _Float16* kp0 = Km + (size_t)(kb + lm) * DHH + 32 * c;
      v16h b0 = make_frag(kp0 + 8 * g, kp0 + 16 + 8 * g);
      s0 = __builtin_amdgcn_wmma_f32_16x16x32_f16(false, aq[c], false, b0,
                                                  (short)0, s0, false, false);
      const _Float16* kp1 = Km + (size_t)(kb + 16 + lm) * DHH + 32 * c;
      v16h b1 = make_frag(kp1 + 8 * g, kp1 + 16 + 8 * g);
      s1 = __builtin_amdgcn_wmma_f32_16x16x32_f16(false, aq[c], false, b1,
                                                  (short)0, s1, false, false);
    }
    // ---- scale + causal mask + online softmax ----
    const int kg0 = kb + lm, kg1 = kb + 16 + lm;
    float mx[8];
#pragma unroll
    for (int r = 0; r < 8; ++r) {
      int qg = q0 + r + 8 * g;
      float x0 = (kg0 <= qg) ? s0[r] * sc : -INFINITY;
      float x1 = (kg1 <= qg) ? s1[r] * sc : -INFINITY;
      s0[r] = x0; s1[r] = x1;
      mx[r] = fmaxf(x0, x1);
    }
#pragma unroll
    for (int m = 1; m <= 8; m <<= 1)
#pragma unroll
      for (int r = 0; r < 8; ++r) mx[r] = fmaxf(mx[r], __shfl_xor(mx[r], m, 32));

    float alpha[8], rs[8];
#pragma unroll
    for (int r = 0; r < 8; ++r) {
      float mn = fmaxf(mrow[r], mx[r]);
      alpha[r] = __expf(mrow[r] - mn);       // exp(-inf)=0 on first live block
      mrow[r]  = mn;
      float p0 = __expf(s0[r] - mn);
      float p1 = __expf(s1[r] - mn);
      s0[r] = p0; s1[r] = p1;
      rs[r] = p0 + p1;
    }
#pragma unroll
    for (int m = 1; m <= 8; m <<= 1)
#pragma unroll
      for (int r = 0; r < 8; ++r) rs[r] += __shfl_xor(rs[r], m, 32);
#pragma unroll
    for (int r = 0; r < 8; ++r) lrow[r] = lrow[r] * alpha[r] + rs[r];
#pragma unroll
    for (int dt = 0; dt < 4; ++dt)
#pragma unroll
      for (int r = 0; r < 8; ++r) out[dt][r] *= alpha[r];

    // ---- bounce P (16x32) through LDS: D layout -> A-fragment layout ----
#pragma unroll
    for (int r = 0; r < 8; ++r) {
      int row = r + 8 * g;
      pbuf[wave][row][lm]      = (_Float16)s0[r];
      pbuf[wave][row][16 + lm] = (_Float16)s1[r];
    }
    asm volatile("s_wait_dscnt 0" ::: "memory");    // wave-local RAW through LDS
    const _Float16* pp = &pbuf[wave][lm][0];
    v16h pa = make_frag(pp + 8 * g, pp + 16 + 8 * g);

    // ---- out[dt] += P @ V   (Vt gives contiguous B-fragment rows) ----
#pragma unroll
    for (int dt = 0; dt < 4; ++dt) {
      const _Float16* vp = Vt + (size_t)(16 * dt + lm) * SS + kb;
      v16h bv = make_frag(vp + 8 * g, vp + 16 + 8 * g);
      out[dt] = __builtin_amdgcn_wmma_f32_16x16x32_f16(false, pa, false, bv,
                                                       (short)0, out[dt], false, false);
    }
  }

  // ---- normalize and emit ctx f16 [B*S, DIM] ----
#pragma unroll
  for (int r = 0; r < 8; ++r) {
    float inv = 1.0f / lrow[r];
    int qg = q0 + r + 8 * g;
#pragma unroll
    for (int dt = 0; dt < 4; ++dt) {
      size_t o = (size_t)(b * SS + qg) * DIMM + h * DHH + 16 * dt + lm;
      ctxh[o] = (_Float16)(out[dt][r] * inv);
    }
  }
}

// ---------------------------------------------------------------------------
// Launch
// ---------------------------------------------------------------------------
extern "C" void kernel_launch(void* const* d_in, const int* in_sizes, int n_in,
                              void* d_out, int out_size, void* d_ws, size_t ws_size,
                              hipStream_t stream) {
  const float* x  = (const float*)d_in[0];
  const float* Wq = (const float*)d_in[1];
  const float* bq = (const float*)d_in[2];
  const float* Wk = (const float*)d_in[3];
  const float* bk = (const float*)d_in[4];
  const float* Wv = (const float*)d_in[5];
  const float* bv = (const float*)d_in[6];
  const float* Wo = (const float*)d_in[7];
  const float* bo = (const float*)d_in[8];
  const float* logit_scale = (const float*)d_in[9];
  // d_in[10] = mask: causal, computed analytically in-kernel.
  float* out = (float*)d_out;

  char* ws = (char*)d_ws;
  // workspace layout (bytes)
  _Float16* xh   = (_Float16*)(ws);                               //  8 MB  [4096,1024]
  _Float16* wcat = (_Float16*)(ws + 8388608);                     //  6 MB  [3072,1024]
  _Float16* woh  = (_Float16*)(ws + 14680064);                    //  2 MB  [1024,1024]
  float*    bcat = (float*)   (ws + 16777216);                    // 12 KB  [3072]
  float*    qkv  = (float*)   (ws + 16793600);                    // 48 MB  [4096,3072]
  _Float16* qh   = (_Float16*)(ws + 67125248);                    //  8 MB  [B,H,S,64]
  _Float16* kh   = (_Float16*)(ws + 75513856);                    //  8 MB
  _Float16* vt   = (_Float16*)(ws + 83902464);                    //  8 MB  [B,H,64,S]
  _Float16* ctxh = (_Float16*)qkv;                                // reuse qkv region (8 MB)

  const int nX = MTOT * DIMM;       // 4 Mi
  const int nW = DIMM * DIMM;       // 1 Mi
  // K0: converts
  cvt_f32_to_f16<<<(nX + 255) / 256, 256, 0, stream>>>(x, xh, nX);
  cvt_f32_to_f16<<<(nW + 255) / 256, 256, 0, stream>>>(Wq, wcat,            nW);
  cvt_f32_to_f16<<<(nW + 255) / 256, 256, 0, stream>>>(Wk, wcat + nW,       nW);
  cvt_f32_to_f16<<<(nW + 255) / 256, 256, 0, stream>>>(Wv, wcat + 2 * nW,   nW);
  cvt_f32_to_f16<<<(nW + 255) / 256, 256, 0, stream>>>(Wo, woh,             nW);
  hipMemcpyAsync(bcat,            bq, DIMM * sizeof(float), hipMemcpyDeviceToDevice, stream);
  hipMemcpyAsync(bcat + DIMM,     bk, DIMM * sizeof(float), hipMemcpyDeviceToDevice, stream);
  hipMemcpyAsync(bcat + 2 * DIMM, bv, DIMM * sizeof(float), hipMemcpyDeviceToDevice, stream);

  // K1: fused QKV projection  [4096,1024] @ [3072,1024]^T (TDM + WMMA)
  gemm_f16_wmma<<<dim3(NQKV / 256, MTOT / 64), 256, 0, stream>>>(
      xh, wcat, bcat, qkv, MTOT, NQKV, DIMM);

  // K2: RoPE + l2norm + V transpose  (B*S*H = 65536 waves, 8 per block)
  rope_norm_kernel<<<(BB * SS * HH) / 8, 256, 0, stream>>>(qkv, qh, kh, vt);

  // K3: flash attention  (grid: B*H x S/64, 4 waves of 16 queries each)
  flash_attn_wmma<<<dim3(BB * HH, SS / 64), 128, 0, stream>>>(qh, kh, vt, logit_scale, ctxh);

  // K4: output projection  [4096,1024] @ [1024,1024]^T + bo -> d_out (f32)
  gemm_f16_wmma<<<dim3(DIMM / 256, MTOT / 64), 256, 0, stream>>>(
      ctxh, woh, bo, out, MTOT, DIMM, DIMM);
}
